// Attention_73796128080520
// MI455X (gfx1250) — matrix operand order repaired and checked by reading, not compiled, not run on hardware
//
#include <hip/hip_runtime.h>
#include <hip/hip_bf16.h>
#include <math.h>

typedef __attribute__((ext_vector_type(16))) _Float16 v16h;
typedef __attribute__((ext_vector_type(8)))  _Float16 v8h;
typedef __attribute__((ext_vector_type(8)))  float    v8f;
typedef __attribute__((ext_vector_type(4)))  float    v4f;

static constexpr int B_   = 128;
static constexpr int N_   = 256;   // 16x16 window
static constexpr int C_   = 512;
static constexpr int H_   = 8;
static constexpr int D_   = 64;    // head dim
static constexpr float SCALE_ = 0.125f; // 64^-0.5

__device__ __forceinline__ v8f wmma_f16(v16h a, v16h b, v8f c) {
    return __builtin_amdgcn_wmma_f32_16x16x32_f16(
        /*neg_a=*/false, a, /*neg_b=*/false, b,
        /*c_mod=*/(short)0, c, /*reuse_a=*/false, /*reuse_b=*/false);
}

// 16 contiguous halves -> v16h  (B operand chunk: lane holds column, K = 16*hi + i)
__device__ __forceinline__ v16h ld16(const _Float16* p) {
    v8h a = *(const v8h*)p;
    v8h b = *(const v8h*)(p + 8);
    v16h r;
#pragma unroll
    for (int i = 0; i < 8; ++i) { r[i] = a[i]; r[i + 8] = b[i]; }
    return r;
}

// A operand from f16 row: lane<16 K={0..15}, lane>=16 K={16..31}
__device__ __forceinline__ v16h ldA16(const _Float16* rowp, int hi) {
    v8h a = *(const v8h*)(rowp + 16 * hi);
    v8h b = *(const v8h*)(rowp + 16 * hi + 8);
    v16h r;
#pragma unroll
    for (int i = 0; i < 8; ++i) { r[i] = a[i]; r[i + 8] = b[i]; }
    return r;
}

// A operand from f32 row, converted to f16 inline
__device__ __forceinline__ v16h ldA32(const float* rowp, int hi) {
    v4f f0 = *(const v4f*)(rowp + 16 * hi);
    v4f f1 = *(const v4f*)(rowp + 16 * hi + 4);
    v4f f2 = *(const v4f*)(rowp + 16 * hi + 8);
    v4f f3 = *(const v4f*)(rowp + 16 * hi + 12);
    v16h r;
#pragma unroll
    for (int i = 0; i < 4; ++i) {
        r[i]      = (_Float16)f0[i];
        r[i + 4]  = (_Float16)f1[i];
        r[i + 8]  = (_Float16)f2[i];
        r[i + 12] = (_Float16)f3[i];
    }
    return r;
}

// async memory -> LDS (16 bytes per lane), tracked by ASYNCcnt
__device__ __forceinline__ void async_g2l_b128(const _Float16* g, _Float16* l) {
    unsigned lds = (unsigned)(unsigned long long)l; // low 32 bits = LDS offset
    asm volatile("global_load_async_to_lds_b128 %0, %1, off"
                 :: "v"(lds), "v"(g) : "memory");
}

// ---------------- kernel 0: transpose+convert weights  w[in,out] f32 -> wt[out,in] f16
__global__ void transpose_w_kernel(const float* __restrict__ w, _Float16* __restrict__ wt,
                                   int in_dim, int out_dim) {
    int i = blockIdx.x * 256 + threadIdx.x;
    int total = in_dim * out_dim;
    if (i < total) {
        int o = i / in_dim;
        int c = i - o * in_dim;
        wt[i] = (_Float16)w[(size_t)c * out_dim + o];
    }
}

// ---------------- kernel 0b: expand relative-position bias -> biasT[h][kcol][qrow] f32 (2 MB)
__global__ void bias_expand_kernel(const float* __restrict__ bt, float* __restrict__ biasT) {
    int i = blockIdx.x * 256 + threadIdx.x;      // over H * 256 * 256
    int h = i >> 16;
    int kq = i & 65535;
    int k = kq >> 8, q = kq & 255;
    int qr = q >> 4, qc = q & 15;
    int kr = k >> 4, kc = k & 15;
    int idx = (qr - kr + 15) * 31 + (qc - kc + 15);
    biasT[i] = bt[idx * H_ + h];
}

// ---------------- kernel 1: q = x@q_w (scaled), kv = y@kv_w; write q,k [B,H,N,d], vT [B,H,d,N] f16
__global__ void qkv_kernel(const float* __restrict__ x, const float* __restrict__ y,
                           const _Float16* __restrict__ qwT, const _Float16* __restrict__ kvwT,
                           _Float16* __restrict__ qh, _Float16* __restrict__ kh,
                           _Float16* __restrict__ vT) {
    const int lane = threadIdx.x & 31;
    const int hi = lane >> 4, n = lane & 15;
    int w = blockIdx.x * 8 + (threadIdx.x >> 5);   // 49152 wave-jobs
    const int QJOBS = 2048 * 8;                     // 2048 m-tiles x 8 col-clusters (q: 512 cols)
    bool is_q = (w < QJOBS);
    int mt, ncl;
    const float* src;
    const _Float16* wt;
    if (is_q) { mt = w >> 3; ncl = w & 7;  src = x; wt = qwT; }
    else      { int w2 = w - QJOBS; mt = w2 >> 4; ncl = w2 & 15; src = y; wt = kvwT; }
    const int c0 = ncl * 64;

    v8f acc[4] = {};
    const float* abase = src + (size_t)mt * 16 * C_ + (size_t)n * C_;
    const _Float16* wb = wt + (size_t)(c0 + n) * C_ + 16 * hi;
#pragma unroll 4
    for (int k0 = 0; k0 < C_; k0 += 32) {
        v16h a  = ldA32(abase + k0, hi);
        v16h b0 = ld16(wb + k0);
        v16h b1 = ld16(wb + 16 * C_ + k0);
        v16h b2 = ld16(wb + 32 * C_ + k0);
        v16h b3 = ld16(wb + 48 * C_ + k0);
        acc[0] = wmma_f16(a, b0, acc[0]);
        acc[1] = wmma_f16(a, b1, acc[1]);
        acc[2] = wmma_f16(a, b2, acc[2]);
        acc[3] = wmma_f16(a, b3, acc[3]);
    }
#pragma unroll
    for (int j = 0; j < 4; ++j) {
        int c = c0 + j * 16 + n;
#pragma unroll
        for (int r = 0; r < 8; ++r) {
            int t = mt * 16 + r + 8 * hi;
            int bb = t >> 8, tok = t & 255;
            float val = acc[j][r];
            if (is_q) {
                int h = c >> 6, dd = c & 63;
                qh[((((size_t)bb * H_ + h) * N_) + tok) * D_ + dd] = (_Float16)(val * SCALE_);
            } else {
                int which = c >> 9, h = (c >> 6) & 7, dd = c & 63;
                if (which == 0)
                    kh[((((size_t)bb * H_ + h) * N_) + tok) * D_ + dd] = (_Float16)val;
                else
                    vT[((((size_t)bb * H_ + h) * D_) + dd) * N_ + tok] = (_Float16)val;
            }
        }
    }
}

// ---------------- kernel 2: flash attention per (b,h). LDS: K 32KB | VT 32KB | Pstage 8KB
__global__ void attn_kernel(const _Float16* __restrict__ qh, const _Float16* __restrict__ kh,
                            const _Float16* __restrict__ vT, const float* __restrict__ biasT,
                            _Float16* __restrict__ attn16) {
    extern __shared__ _Float16 smem[];
    const int bh = blockIdx.x;
    const int bb = bh >> 3, h = bh & 7;
    const int tid = threadIdx.x, lane = tid & 31, wave = tid >> 5;
    const int hi = lane >> 4, n = lane & 15;

    _Float16* sK = smem;                       // [256][64]
    _Float16* sV = smem + 16384;               // [64][256]  (V transposed)
    _Float16* sP = smem + 32768 + wave * 512;  // per-wave 16x32 staging

    // async DMA of K and V^T tiles straight into LDS (ASYNCcnt tracked)
    const _Float16* gk = kh + (size_t)bh * (N_ * D_);
    const _Float16* gv = vT + (size_t)bh * (N_ * D_);
    for (int i = tid; i < (N_ * D_) / 8; i += 256) {
        async_g2l_b128(gk + (size_t)i * 8, sK + (size_t)i * 8);
        async_g2l_b128(gv + (size_t)i * 8, sV + (size_t)i * 8);
    }
    asm volatile("s_wait_asynccnt 0" ::: "memory");
    __syncthreads();

    const float* bslice = biasT + (size_t)h * N_ * N_;

    for (int qt = wave; qt < 16; qt += 8) {
        const _Float16* qrow = qh + (size_t)bh * (N_ * D_) + (size_t)(qt * 16 + n) * D_;
        v16h aq0 = ldA16(qrow, hi);
        v16h aq1 = ldA16(qrow + 32, hi);

        v8f O0 = {}, O1 = {}, O2 = {}, O3 = {};
        float m[8], l[8];
#pragma unroll
        for (int r = 0; r < 8; ++r) { m[r] = -INFINITY; l[r] = 0.f; }

        for (int kk = 0; kk < 8; ++kk) {
            v8f s0, s1;
#pragma unroll
            for (int half = 0; half < 2; ++half) {
                int kt = kk * 2 + half;
                const _Float16* kb = sK + (size_t)(kt * 16 + n) * D_;
                v16h b0 = ld16(kb + 16 * hi);
                v16h b1 = ld16(kb + 32 + 16 * hi);
                v8f sv = {};
                sv = wmma_f16(aq0, b0, sv);
                sv = wmma_f16(aq1, b1, sv);
                // relative position bias: contiguous in qrow for this lane's 8 rows
                const float* bp = bslice + (size_t)(kt * 16 + n) * N_ + qt * 16 + 8 * hi;
                v4f bA = *(const v4f*)bp;
                v4f bB = *(const v4f*)(bp + 4);
#pragma unroll
                for (int r = 0; r < 4; ++r) { sv[r] += bA[r]; sv[r + 4] += bB[r]; }
                if (half == 0) s0 = sv; else s1 = sv;
            }
            // online softmax over the 32 new columns
            float corr[8];
#pragma unroll
            for (int r = 0; r < 8; ++r) {
                float tm = fmaxf(s0[r], s1[r]);
                tm = fmaxf(tm, __shfl_xor(tm, 1));
                tm = fmaxf(tm, __shfl_xor(tm, 2));
                tm = fmaxf(tm, __shfl_xor(tm, 4));
                tm = fmaxf(tm, __shfl_xor(tm, 8));
                float mn = fmaxf(m[r], tm);
                corr[r] = __expf(m[r] - mn);
                float p0 = __expf(s0[r] - mn);
                float p1 = __expf(s1[r] - mn);
                float ps = p0 + p1;
                ps += __shfl_xor(ps, 1);
                ps += __shfl_xor(ps, 2);
                ps += __shfl_xor(ps, 4);
                ps += __shfl_xor(ps, 8);
                l[r] = l[r] * corr[r] + ps;
                m[r] = mn;
                sP[(r + 8 * hi) * 32 + n]      = (_Float16)p0;
                sP[(r + 8 * hi) * 32 + 16 + n] = (_Float16)p1;
            }
#pragma unroll
            for (int r = 0; r < 8; ++r) {
                O0[r] *= corr[r]; O1[r] *= corr[r]; O2[r] *= corr[r]; O3[r] *= corr[r];
            }
            // P staging (C-layout stores) must be visible to this wave's A-layout reads
            asm volatile("s_wait_dscnt 0" ::: "memory");
            v16h ap = ldA16(sP + n * 32, hi);
            const _Float16* vb = sV + (size_t)n * N_ + kk * 32 + 16 * hi;
            v16h bv0 = ld16(vb);
            v16h bv1 = ld16(vb + 16 * N_);
            v16h bv2 = ld16(vb + 32 * N_);
            v16h bv3 = ld16(vb + 48 * N_);
            O0 = wmma_f16(ap, bv0, O0);
            O1 = wmma_f16(ap, bv1, O1);
            O2 = wmma_f16(ap, bv2, O2);
            O3 = wmma_f16(ap, bv3, O3);
        }
        // epilogue: divide by l, store f16 [B*N, C] with col = h*64 + dd
#pragma unroll
        for (int r = 0; r < 8; ++r) {
            float inv = 1.0f / l[r];
            size_t tok = (size_t)bb * N_ + qt * 16 + r + 8 * hi;
            _Float16* orow = attn16 + tok * C_ + h * D_;
            orow[0 * 16 + n] = (_Float16)(O0[r] * inv);
            orow[1 * 16 + n] = (_Float16)(O1[r] * inv);
            orow[2 * 16 + n] = (_Float16)(O2[r] * inv);
            orow[3 * 16 + n] = (_Float16)(O3[r] * inv);
        }
    }
}

// ---------------- kernel 3: out = attn16 @ proj_wT + proj_b  (f32 out)
__global__ void proj_kernel(const _Float16* __restrict__ attn16, const _Float16* __restrict__ pwT,
                            const float* __restrict__ pb, float* __restrict__ out) {
    const int lane = threadIdx.x & 31;
    const int hi = lane >> 4, n = lane & 15;
    int w = blockIdx.x * 8 + (threadIdx.x >> 5); // 16384 jobs
    int mt = w >> 3, ncl = w & 7;
    int c0 = ncl * 64;

    v8f acc[4] = {};
    const _Float16* abase = attn16 + (size_t)mt * 16 * C_ + (size_t)n * C_;
    const _Float16* wb = pwT + (size_t)(c0 + n) * C_ + 16 * hi;
#pragma unroll 4
    for (int k0 = 0; k0 < C_; k0 += 32) {
        v16h a  = ldA16(abase + k0, hi);
        v16h b0 = ld16(wb + k0);
        v16h b1 = ld16(wb + 16 * C_ + k0);
        v16h b2 = ld16(wb + 32 * C_ + k0);
        v16h b3 = ld16(wb + 48 * C_ + k0);
        acc[0] = wmma_f16(a, b0, acc[0]);
        acc[1] = wmma_f16(a, b1, acc[1]);
        acc[2] = wmma_f16(a, b2, acc[2]);
        acc[3] = wmma_f16(a, b3, acc[3]);
    }
#pragma unroll
    for (int j = 0; j < 4; ++j) {
        int c = c0 + j * 16 + n;
        float bias = pb[c];
#pragma unroll
        for (int r = 0; r < 8; ++r) {
            size_t t = (size_t)mt * 16 + r + 8 * hi;
            out[t * C_ + c] = acc[j][r] + bias;
        }
    }
}

extern "C" void kernel_launch(void* const* d_in, const int* in_sizes, int n_in,
                              void* d_out, int out_size, void* d_ws, size_t ws_size,
                              hipStream_t stream) {
    const float* x      = (const float*)d_in[0];
    const float* y      = (const float*)d_in[1];
    const float* q_w    = (const float*)d_in[2];
    const float* kv_w   = (const float*)d_in[3];
    const float* proj_w = (const float*)d_in[4];
    const float* proj_b = (const float*)d_in[5];
    const float* bias_t = (const float*)d_in[6];
    float* out = (float*)d_out;

    char* ws = (char*)d_ws;
    size_t off = 0;
    const size_t QKV_BYTES = (size_t)B_ * H_ * N_ * D_ * sizeof(_Float16); // 32 MiB
    _Float16* qh     = (_Float16*)(ws + off); off += QKV_BYTES;
    _Float16* kh     = (_Float16*)(ws + off); off += QKV_BYTES;
    _Float16* vT     = (_Float16*)(ws + off); off += QKV_BYTES;
    _Float16* attn16 = (_Float16*)(ws + off); off += (size_t)B_ * N_ * C_ * sizeof(_Float16);
    _Float16* qwT    = (_Float16*)(ws + off); off += (size_t)C_ * C_ * sizeof(_Float16);
    _Float16* kvwT   = (_Float16*)(ws + off); off += (size_t)C_ * 2 * C_ * sizeof(_Float16);
    _Float16* pwT    = (_Float16*)(ws + off); off += (size_t)C_ * C_ * sizeof(_Float16);
    float*    biasT  = (float*)(ws + off);    off += (size_t)H_ * N_ * N_ * sizeof(float);

    // 0) weight prep (transpose + f16) and bias field expansion
    transpose_w_kernel<<<(C_ * C_ + 255) / 256, 256, 0, stream>>>(q_w, qwT, C_, C_);
    transpose_w_kernel<<<(C_ * 2 * C_ + 255) / 256, 256, 0, stream>>>(kv_w, kvwT, C_, 2 * C_);
    transpose_w_kernel<<<(C_ * C_ + 255) / 256, 256, 0, stream>>>(proj_w, pwT, C_, C_);
    bias_expand_kernel<<<(H_ * N_ * N_) / 256, 256, 0, stream>>>(bias_t, biasT);

    // 1) q/k/v projection: 2048 m-tiles * (8 q + 16 kv) col-clusters = 49152 wave jobs / 8 waves
    qkv_kernel<<<6144, 256, 0, stream>>>(x, y, qwT, kvwT, qh, kh, vT);

    // 2) attention: one block per (b,h); 72 KB dynamic LDS
    attn_kernel<<<B_ * H_, 256, 73728, stream>>>(qh, kh, vT, biasT, attn16);

    // 3) output projection: 2048 m-tiles * 8 clusters = 16384 wave jobs / 8 waves
    proj_kernel<<<2048, 256, 0, stream>>>(attn16, pwT, proj_b, out);
}
